// RouteNet_49520972922897
// MI455X (gfx1250) — matrix operand order repaired and checked by reading, not compiled, not run on hardware
//
#include <hip/hip_runtime.h>

#define P_N 100000
#define L_N 20000
#define T_N 8
#define DEG_N 48
#define D_N 64
#define ITERS_N 4
#define WAVES 4
#define NTP (P_N / 16)   // 6250 path tiles
#define NTL (L_N / 16)   // 1250 link tiles

typedef __attribute__((ext_vector_type(16))) _Float16 v16h;
typedef __attribute__((ext_vector_type(8)))  _Float16 v8h;
typedef __attribute__((ext_vector_type(8)))  float    v8f;
typedef __attribute__((ext_vector_type(2)))  __fp16   v2fp16;

__device__ __forceinline__ v16h join16(v8h a, v8h b) {
  return __builtin_shufflevector(a, b, 0,1,2,3,4,5,6,7,8,9,10,11,12,13,14,15);
}

__device__ __forceinline__ v8f wmma_f16(v16h a, v16h b, v8f c) {
  // D(16x16,f32) += A(16x32,f16) * B(32x16,f16)
  return __builtin_amdgcn_wmma_f32_16x16x32_f16(false, a, false, b, (short)0, c,
                                                false, false);
}

// A fragment from an f16 row-major [16][Kt] LDS tile, K-chunk c (32 wide).
// Lane l: M = l&15, halves k = c*32 + 8*(l>=16) + {0..7} and +16.
__device__ __forceinline__ v16h load_a_frag(const _Float16* buf, int lane, int Kt, int c) {
  const int m   = lane & 15;
  const int khi = ((lane >> 4) & 1) << 3;   // 0 or 8 halves
  const _Float16* p = buf + m * Kt + c * 32 + khi;
  v8h lo = *(const v8h*)(p);
  v8h hi = *(const v8h*)(p + 16);
  return join16(lo, hi);
}

// B fragment: W is row-major [N][K] f16 in global memory; B[k][n] = W[n][k].
// Lane l: N = (l&15)+16*jn, halves k = c*32 + 8*(l>=16) + {0..7} and +16.
__device__ __forceinline__ v16h load_b_frag(const _Float16* W, int lane, int K, int jn, int c) {
  const int n   = (lane & 15) + (jn << 4);
  const int khi = ((lane >> 4) & 1) << 3;   // 0 or 8 halves
  const _Float16* p = W + n * K + c * 32 + khi;
  v8h lo = *(const v8h*)(p);
  v8h hi = *(const v8h*)(p + 16);
  return join16(lo, hi);
}

// Fast activations: v_rcp_f32 / v_exp_f32 / v_tanh_f32 are TRANS ops that
// co-execute with XDL WMMA on gfx1250 — avoid IEEE division chains.
__device__ __forceinline__ float fast_sigmoid(float x) {
  return __builtin_amdgcn_rcpf(1.0f + __expf(-x));
}
#if defined(__has_builtin)
#if __has_builtin(__builtin_amdgcn_tanhf)
#define FAST_TANH(x) __builtin_amdgcn_tanhf(x)
#elif __has_builtin(__builtin_amdgcn_tanh_f32)
#define FAST_TANH(x) __builtin_amdgcn_tanh_f32(x)
#endif
#endif
#ifndef FAST_TANH
#define FAST_TANH(x) (1.0f - 2.0f * __builtin_amdgcn_rcpf(__expf(2.0f * (x)) + 1.0f))
#endif

// Packed f32x2 -> f16x2 convert + single 32-bit store (v_cvt_pk_rtz_f16_f32).
__device__ __forceinline__ void store_h2(_Float16* dst, float x, float y) {
  *(v2fp16*)dst = __builtin_amdgcn_cvt_pkrtz(x, y);
}
__device__ __forceinline__ void store_h4(_Float16* dst, float4 v) {
  store_h2(dst,     v.x, v.y);
  store_h2(dst + 2, v.z, v.w);
}

// ---------------------------------------------------------------------------
// f32 -> f16 weight conversion (keeps weights L2-resident in WMMA-friendly form)
// ---------------------------------------------------------------------------
__global__ void cvt_f16_kernel(const float* __restrict__ src, _Float16* __restrict__ dst, int n) {
  int i = blockIdx.x * blockDim.x + threadIdx.x;
  if (i < n) dst[i] = (_Float16)src[i];
}

// ---------------------------------------------------------------------------
// 2-layer embedding MLP: out = relu(relu(x*w1 + b1) @ w2^T + b2), x scalar/row
// ---------------------------------------------------------------------------
__global__ __launch_bounds__(256) void embed_kernel(
    const float* __restrict__ x, int n,
    const float* __restrict__ w1, const float* __restrict__ b1,
    const float* __restrict__ w2, const float* __restrict__ b2,
    float* __restrict__ out)
{
  __shared__ float w2s[D_N * D_N];
  for (int i = threadIdx.x; i < D_N * D_N; i += blockDim.x) w2s[i] = w2[i];
  __syncthreads();
  int row = blockIdx.x * blockDim.x + threadIdx.x;
  if (row >= n) return;
  float t = x[row];
  float h1[D_N];
  #pragma unroll
  for (int d = 0; d < D_N; ++d) h1[d] = fmaxf(t * w1[d] + b1[d], 0.0f);
  for (int o = 0; o < D_N; ++o) {
    float a = b2[o];
    #pragma unroll
    for (int d = 0; d < D_N; ++d) a += h1[d] * w2s[o * D_N + d];
    out[(size_t)row * D_N + o] = fmaxf(a, 0.0f);
  }
}

// ---------------------------------------------------------------------------
// Path GRU scan: one wave per 16-path tile, T=8 steps, WMMA gate GEMMs.
// Writes pss[p][0]=h0, pss[p][t+1]=out_t, path_state=h_final.
// ---------------------------------------------------------------------------
__global__ __launch_bounds__(128) void gru_path_kernel(
    const float* __restrict__ link_state,
    float* __restrict__ path_state,
    float* __restrict__ pss,
    const int* __restrict__ l2p,              // [P][T]
    const _Float16* __restrict__ wih16,       // [192][64]
    const _Float16* __restrict__ whh16,       // [192][64]
    const float* __restrict__ bih,            // [192]
    const float* __restrict__ bhh)            // [192]
{
  __shared__ alignas(16) _Float16 xs[WAVES][16 * D_N];
  __shared__ alignas(16) _Float16 hsbuf[WAVES][16 * D_N];
  __shared__ int vli[WAVES][16];

  const int wave = threadIdx.x >> 5;
  const int lane = threadIdx.x & 31;
  const int gtile = blockIdx.x * WAVES + wave;
  const bool writer = gtile < NTP;             // clamped tail keeps EXEC all-ones
  const int tile = writer ? gtile : (NTP - 1);
  const int pbase = tile * 16;

  const int m2 = lane >> 1;
  const int half = lane & 1;
  const int n0 = lane & 15;
  const int mh = ((lane >> 4) & 1) << 3;

  // Stage h0 (f16) and write pss[:,0,:]
  {
    const float* src = path_state + (size_t)(pbase + m2) * D_N + half * 32;
    float* dpss = pss + (size_t)(pbase + m2) * ((T_N + 1) * D_N) + half * 32;
    _Float16* hb = &hsbuf[wave][m2 * D_N + half * 32];
    #pragma unroll
    for (int c4 = 0; c4 < 8; ++c4) {
      float4 v = ((const float4*)src)[c4];
      if (writer) ((float4*)dpss)[c4] = v;
      store_h4(hb + c4 * 4, v);
    }
  }
  __syncthreads();

  float hreg[4][8];
  #pragma unroll
  for (int j = 0; j < 4; ++j)
    #pragma unroll
    for (int r = 0; r < 8; ++r)
      hreg[j][r] = (float)hsbuf[wave][(r + mh) * D_N + n0 + 16 * j];

  for (int t = 0; t < T_N; ++t) {
    // Stage x_t = link_state[l2p[p][t]] (or 0), two lanes per path row.
    int li = l2p[(pbase + m2) * T_N + t];
    if (half == 0) vli[wave][m2] = li;
    {
      _Float16* xb = &xs[wave][m2 * D_N + half * 32];
      if (li >= 0) {
        const float* src = link_state + (size_t)li * D_N + half * 32;
        #pragma unroll
        for (int c4 = 0; c4 < 8; ++c4)
          store_h4(xb + c4 * 4, ((const float4*)src)[c4]);
      } else {
        #pragma unroll
        for (int i = 0; i < 16; ++i) store_h2(xb + i * 2, 0.0f, 0.0f);
      }
    }
    __syncthreads();

    int vmask[8];
    #pragma unroll
    for (int r = 0; r < 8; ++r) vmask[r] = vli[wave][r + mh];

    v16h ax0 = load_a_frag(&xs[wave][0], lane, D_N, 0);
    v16h ax1 = load_a_frag(&xs[wave][0], lane, D_N, 1);
    v16h ah0 = load_a_frag(&hsbuf[wave][0], lane, D_N, 0);
    v16h ah1 = load_a_frag(&hsbuf[wave][0], lane, D_N, 1);

    // r and z gates need only (gi + gh): chain all 4 WMMAs into ONE
    // accumulator seeded with bih+bhh (halves live accumulators).
    float rg[4][8], zg[4][8];
    #pragma unroll
    for (int g = 0; g < 2; ++g) {
      #pragma unroll
      for (int j2 = 0; j2 < 4; ++j2) {
        const int j = g * 4 + j2;
        const float b0 = bih[16 * j + n0] + bhh[16 * j + n0];
        v8f acc;
        #pragma unroll
        for (int r = 0; r < 8; ++r) acc[r] = b0;
        acc = wmma_f16(ax0, load_b_frag(wih16, lane, D_N, j, 0), acc);
        acc = wmma_f16(ax1, load_b_frag(wih16, lane, D_N, j, 1), acc);
        acc = wmma_f16(ah0, load_b_frag(whh16, lane, D_N, j, 0), acc);
        acc = wmma_f16(ah1, load_b_frag(whh16, lane, D_N, j, 1), acc);
        #pragma unroll
        for (int r = 0; r < 8; ++r) {
          float s = fast_sigmoid(acc[r]);
          if (g == 0) rg[j2][r] = s; else zg[j2][r] = s;
        }
      }
    }
    // n gate: needs inn and hn separately (hn scaled by r before the add)
    #pragma unroll
    for (int j2 = 0; j2 < 4; ++j2) {
      const int j = 8 + j2;
      const float bi = bih[16 * j + n0];
      const float bh = bhh[16 * j + n0];
      v8f gi, gh;
      #pragma unroll
      for (int r = 0; r < 8; ++r) { gi[r] = bi; gh[r] = bh; }
      gi = wmma_f16(ax0, load_b_frag(wih16, lane, D_N, j, 0), gi);
      gi = wmma_f16(ax1, load_b_frag(wih16, lane, D_N, j, 1), gi);
      gh = wmma_f16(ah0, load_b_frag(whh16, lane, D_N, j, 0), gh);
      gh = wmma_f16(ah1, load_b_frag(whh16, lane, D_N, j, 1), gh);
      #pragma unroll
      for (int r = 0; r < 8; ++r) {
        float ng = FAST_TANH(gi[r] + rg[j2][r] * gh[r]);
        float hn = (1.0f - zg[j2][r]) * ng + zg[j2][r] * hreg[j2][r];
        bool valid = vmask[r] >= 0;
        float hnew = valid ? hn : hreg[j2][r];
        float outv = valid ? hn : 0.0f;
        hreg[j2][r] = hnew;
        if (writer)
          pss[(size_t)(pbase + r + mh) * ((T_N + 1) * D_N) +
              (size_t)(t + 1) * D_N + n0 + 16 * j2] = outv;
      }
    }
    __syncthreads();
    // Stage updated h for next step's gh A-fragments.
    #pragma unroll
    for (int j2 = 0; j2 < 4; ++j2)
      #pragma unroll
      for (int r = 0; r < 8; ++r)
        hsbuf[wave][(r + mh) * D_N + n0 + 16 * j2] = (_Float16)hreg[j2][r];
    __syncthreads();
  }

  if (writer) {
    #pragma unroll
    for (int j2 = 0; j2 < 4; ++j2)
      #pragma unroll
      for (int r = 0; r < 8; ++r)
        path_state[(size_t)(pbase + r + mh) * D_N + n0 + 16 * j2] = hreg[j2][r];
  }
}

// ---------------------------------------------------------------------------
// Link update: gather-reduce over DEG, 3-layer MLP (WMMA), GRU cell (WMMA).
// One wave per 16-link tile. In-place link_state update.
// ---------------------------------------------------------------------------
__global__ __launch_bounds__(128) void link_update_kernel(
    const float* __restrict__ pss,
    float* __restrict__ link_state,
    const int* __restrict__ p2l,              // [L][DEG][2]
    const _Float16* __restrict__ w1, const float* __restrict__ b1,   // [128][256]
    const _Float16* __restrict__ w2, const float* __restrict__ b2,   // [128][128]
    const _Float16* __restrict__ w3, const float* __restrict__ b3,   // [64][128]
    const _Float16* __restrict__ cwih, const _Float16* __restrict__ cwhh, // [192][64]
    const float* __restrict__ cbih, const float* __restrict__ cbhh)
{
  __shared__ alignas(16) _Float16 bufA[WAVES][16 * 256];
  __shared__ alignas(16) _Float16 bufB[WAVES][16 * 128];

  const int wave = threadIdx.x >> 5;
  const int lane = threadIdx.x & 31;
  const int gtile = blockIdx.x * WAVES + wave;
  const bool writer = gtile < NTL;
  const int tile = writer ? gtile : (NTL - 1);
  const int lbase = tile * 16;

  const int m2 = lane >> 1;
  const int half = lane & 1;
  const int n0 = lane & 15;
  const int mh = ((lane >> 4) & 1) << 3;

  // 1) gather + masked min/max/sum/mean over DEG (two lanes per link row)
  const float INF = __builtin_inff();
  float vmin[32], vmax[32], vsum[32];
  #pragma unroll
  for (int i = 0; i < 32; ++i) { vmin[i] = INF; vmax[i] = -INF; vsum[i] = 0.0f; }
  int cnt = 0;
  for (int e = 0; e < DEG_N; ++e) {
    int base = ((lbase + m2) * DEG_N + e) * 2;
    int p0 = p2l[base + 0];
    int p1 = p2l[base + 1];
    if (p0 >= 0) {
      ++cnt;
      const float* src = pss + ((size_t)p0 * (T_N + 1) + (size_t)p1) * D_N + half * 32;
      #pragma unroll
      for (int c4 = 0; c4 < 8; ++c4) {
        float4 v = ((const float4*)src)[c4];
        float vv[4] = {v.x, v.y, v.z, v.w};
        #pragma unroll
        for (int q = 0; q < 4; ++q) {
          int i = c4 * 4 + q;
          vmin[i] = fminf(vmin[i], vv[q]);
          vmax[i] = fmaxf(vmax[i], vv[q]);
          vsum[i] += vv[q];
        }
      }
    }
  }
  {
    float rc = __builtin_amdgcn_rcpf((float)(cnt > 0 ? cnt : 1));
    _Float16* ab = &bufA[wave][m2 * 256];
    #pragma unroll
    for (int i = 0; i < 32; i += 2) {
      int d = half * 32 + i;
      store_h2(&ab[0 * D_N + d], vmin[i], vmin[i + 1]);
      store_h2(&ab[1 * D_N + d], vmax[i], vmax[i + 1]);
      store_h2(&ab[2 * D_N + d], vsum[i], vsum[i + 1]);
      store_h2(&ab[3 * D_N + d], vsum[i] * rc, vsum[i + 1] * rc);
    }
  }
  __syncthreads();

  // 2) layer1: [16,256] @ [256,128] -> relu -> bufB (stride 128)
  #pragma unroll
  for (int j = 0; j < 8; ++j) {
    float bb = b1[16 * j + n0];
    v8f acc;
    #pragma unroll
    for (int r = 0; r < 8; ++r) acc[r] = bb;
    #pragma unroll
    for (int c = 0; c < 8; ++c)
      acc = wmma_f16(load_a_frag(&bufA[wave][0], lane, 256, c),
                     load_b_frag(w1, lane, 256, j, c), acc);
    #pragma unroll
    for (int r = 0; r < 8; ++r)
      bufB[wave][(r + mh) * 128 + n0 + 16 * j] = (_Float16)fmaxf(acc[r], 0.0f);
  }
  __syncthreads();

  // 3) layer2: [16,128] @ [128,128] -> relu -> bufA (stride 128)
  #pragma unroll
  for (int j = 0; j < 8; ++j) {
    float bb = b2[16 * j + n0];
    v8f acc;
    #pragma unroll
    for (int r = 0; r < 8; ++r) acc[r] = bb;
    #pragma unroll
    for (int c = 0; c < 4; ++c)
      acc = wmma_f16(load_a_frag(&bufB[wave][0], lane, 128, c),
                     load_b_frag(w2, lane, 128, j, c), acc);
    #pragma unroll
    for (int r = 0; r < 8; ++r)
      bufA[wave][(r + mh) * 128 + n0 + 16 * j] = (_Float16)fmaxf(acc[r], 0.0f);
  }
  __syncthreads();

  // 4) layer3: [16,128] @ [128,64] -> relu -> pa in bufB[0:1024] (stride 64);
  //    stage old link h (f16) into bufB[1024:2048].
  #pragma unroll
  for (int j = 0; j < 4; ++j) {
    float bb = b3[16 * j + n0];
    v8f acc;
    #pragma unroll
    for (int r = 0; r < 8; ++r) acc[r] = bb;
    #pragma unroll
    for (int c = 0; c < 4; ++c)
      acc = wmma_f16(load_a_frag(&bufA[wave][0], lane, 128, c),
                     load_b_frag(w3, lane, 128, j, c), acc);
    #pragma unroll
    for (int r = 0; r < 8; ++r)
      bufB[wave][(r + mh) * D_N + n0 + 16 * j] = (_Float16)fmaxf(acc[r], 0.0f);
  }
  {
    const float* src = link_state + (size_t)(lbase + m2) * D_N + half * 32;
    _Float16* hb = &bufB[wave][1024 + m2 * D_N + half * 32];
    #pragma unroll
    for (int c4 = 0; c4 < 8; ++c4)
      store_h4(hb + c4 * 4, ((const float4*)src)[c4]);
  }
  __syncthreads();

  // 5) GRU cell: link_state = GRU(pa, h)
  v16h ap0 = load_a_frag(&bufB[wave][0], lane, D_N, 0);
  v16h ap1 = load_a_frag(&bufB[wave][0], lane, D_N, 1);
  v16h ah0 = load_a_frag(&bufB[wave][1024], lane, D_N, 0);
  v16h ah1 = load_a_frag(&bufB[wave][1024], lane, D_N, 1);

  float hreg[4][8];
  #pragma unroll
  for (int j2 = 0; j2 < 4; ++j2)
    #pragma unroll
    for (int r = 0; r < 8; ++r)
      hreg[j2][r] = (float)bufB[wave][1024 + (r + mh) * D_N + n0 + 16 * j2];

  float rg[4][8], zg[4][8];
  #pragma unroll
  for (int g = 0; g < 2; ++g) {
    #pragma unroll
    for (int j2 = 0; j2 < 4; ++j2) {
      const int j = g * 4 + j2;
      const float b0 = cbih[16 * j + n0] + cbhh[16 * j + n0];
      v8f acc;
      #pragma unroll
      for (int r = 0; r < 8; ++r) acc[r] = b0;
      acc = wmma_f16(ap0, load_b_frag(cwih, lane, D_N, j, 0), acc);
      acc = wmma_f16(ap1, load_b_frag(cwih, lane, D_N, j, 1), acc);
      acc = wmma_f16(ah0, load_b_frag(cwhh, lane, D_N, j, 0), acc);
      acc = wmma_f16(ah1, load_b_frag(cwhh, lane, D_N, j, 1), acc);
      #pragma unroll
      for (int r = 0; r < 8; ++r) {
        float s = fast_sigmoid(acc[r]);
        if (g == 0) rg[j2][r] = s; else zg[j2][r] = s;
      }
    }
  }
  #pragma unroll
  for (int j2 = 0; j2 < 4; ++j2) {
    const int j = 8 + j2;
    const float bi = cbih[16 * j + n0];
    const float bh = cbhh[16 * j + n0];
    v8f gi, gh;
    #pragma unroll
    for (int r = 0; r < 8; ++r) { gi[r] = bi; gh[r] = bh; }
    gi = wmma_f16(ap0, load_b_frag(cwih, lane, D_N, j, 0), gi);
    gi = wmma_f16(ap1, load_b_frag(cwih, lane, D_N, j, 1), gi);
    gh = wmma_f16(ah0, load_b_frag(cwhh, lane, D_N, j, 0), gh);
    gh = wmma_f16(ah1, load_b_frag(cwhh, lane, D_N, j, 1), gh);
    #pragma unroll
    for (int r = 0; r < 8; ++r) {
      float ng = FAST_TANH(gi[r] + rg[j2][r] * gh[r]);
      float hn = (1.0f - zg[j2][r]) * ng + zg[j2][r] * hreg[j2][r];
      if (writer)
        link_state[(size_t)(lbase + r + mh) * D_N + n0 + 16 * j2] = hn;
    }
  }
}

// ---------------------------------------------------------------------------
// Host-side orchestration
// ---------------------------------------------------------------------------
extern "C" void kernel_launch(void* const* d_in, const int* in_sizes, int n_in,
                              void* d_out, int out_size, void* d_ws, size_t ws_size,
                              hipStream_t stream)
{
  const float* traffic  = (const float*)d_in[0];
  const float* capacity = (const float*)d_in[1];
  const int*   l2p      = (const int*)d_in[2];
  const int*   p2l      = (const int*)d_in[3];
  const float* pe_w1 = (const float*)d_in[4];
  const float* pe_b1 = (const float*)d_in[5];
  const float* pe_w2 = (const float*)d_in[6];
  const float* pe_b2 = (const float*)d_in[7];
  const float* le_w1 = (const float*)d_in[8];
  const float* le_b1 = (const float*)d_in[9];
  const float* le_w2 = (const float*)d_in[10];
  const float* le_b2 = (const float*)d_in[11];
  const float* gru_wih = (const float*)d_in[12];
  const float* gru_whh = (const float*)d_in[13];
  const float* gru_bih = (const float*)d_in[14];
  const float* gru_bhh = (const float*)d_in[15];
  const float* cell_wih = (const float*)d_in[16];
  const float* cell_whh = (const float*)d_in[17];
  const float* cell_bih = (const float*)d_in[18];
  const float* cell_bhh = (const float*)d_in[19];
  const float* am_w1 = (const float*)d_in[20];
  const float* am_b1 = (const float*)d_in[21];
  const float* am_w2 = (const float*)d_in[22];
  const float* am_b2 = (const float*)d_in[23];
  const float* am_w3 = (const float*)d_in[24];
  const float* am_b3 = (const float*)d_in[25];

  char* ws = (char*)d_ws;
  size_t off = 0;
  float* path_state = (float*)(ws + off); off += (size_t)P_N * D_N * sizeof(float);
  float* link_state = (float*)(ws + off); off += (size_t)L_N * D_N * sizeof(float);
  float* pss        = (float*)(ws + off); off += (size_t)P_N * (T_N + 1) * D_N * sizeof(float);
  _Float16* wih16  = (_Float16*)(ws + off); off += (size_t)3 * D_N * D_N * sizeof(_Float16);
  _Float16* whh16  = (_Float16*)(ws + off); off += (size_t)3 * D_N * D_N * sizeof(_Float16);
  _Float16* cwih16 = (_Float16*)(ws + off); off += (size_t)3 * D_N * D_N * sizeof(_Float16);
  _Float16* cwhh16 = (_Float16*)(ws + off); off += (size_t)3 * D_N * D_N * sizeof(_Float16);
  _Float16* w1_16  = (_Float16*)(ws + off); off += (size_t)128 * 256 * sizeof(_Float16);
  _Float16* w2_16  = (_Float16*)(ws + off); off += (size_t)128 * 128 * sizeof(_Float16);
  _Float16* w3_16  = (_Float16*)(ws + off); off += (size_t)64 * 128 * sizeof(_Float16);

  // Weight conversion to f16 (row-major, WMMA B-fragment friendly)
  auto cvt = [&](const float* s, _Float16* d, int n) {
    cvt_f16_kernel<<<(n + 255) / 256, 256, 0, stream>>>(s, d, n);
  };
  cvt(gru_wih,  wih16,  3 * D_N * D_N);
  cvt(gru_whh,  whh16,  3 * D_N * D_N);
  cvt(cell_wih, cwih16, 3 * D_N * D_N);
  cvt(cell_whh, cwhh16, 3 * D_N * D_N);
  cvt(am_w1,    w1_16,  128 * 256);
  cvt(am_w2,    w2_16,  128 * 128);
  cvt(am_w3,    w3_16,  64 * 128);

  // Embedding MLPs
  embed_kernel<<<(P_N + 255) / 256, 256, 0, stream>>>(traffic, P_N, pe_w1, pe_b1, pe_w2, pe_b2, path_state);
  embed_kernel<<<(L_N + 255) / 256, 256, 0, stream>>>(capacity, L_N, le_w1, le_b1, le_w2, le_b2, link_state);

  // Message-passing iterations
  const int gridA = (NTP + WAVES - 1) / WAVES;   // 1563
  const int gridB = (NTL + WAVES - 1) / WAVES;   // 313
  for (int it = 0; it < ITERS_N; ++it) {
    gru_path_kernel<<<gridA, 128, 0, stream>>>(link_state, path_state, pss, l2p,
                                               wih16, whh16, gru_bih, gru_bhh);
    link_update_kernel<<<gridB, 128, 0, stream>>>(pss, link_state, p2l,
                                                  w1_16, am_b1, w2_16, am_b2, w3_16, am_b3,
                                                  cwih16, cwhh16, cell_bih, cell_bhh);
  }

  // Outputs: [path_state (P*D), link_state (L*D)]
  hipMemcpyAsync(d_out, path_state, (size_t)P_N * D_N * sizeof(float),
                 hipMemcpyDeviceToDevice, stream);
  hipMemcpyAsync((float*)d_out + (size_t)P_N * D_N, link_state,
                 (size_t)L_N * D_N * sizeof(float), hipMemcpyDeviceToDevice, stream);
}
